// DeformableTransformer_12335146074356
// MI455X (gfx1250) — matrix operand order repaired and checked by reading, not compiled, not run on hardware
//
#include <hip/hip_runtime.h>

// ---------------- problem constants ----------------
constexpr int BB   = 2;
constexpr int CDIM = 256;
constexpr int NHD  = 8;
constexpr int HDIM = 32;     // == wave32: one lane per channel in sampling kernels
constexpr int NLV  = 4;
constexpr int NPT  = 4;
constexpr int NLAY = 6;
constexpr int NQRY = 300;
constexpr int FFD  = 1024;
constexpr int LENT = 13294;          // 100^2 + 50^2 + 25^2 + 13^2
constexpr int ME   = BB * LENT;      // 26588 encoder rows
constexpr int MD   = BB * NQRY;      // 600 decoder rows
constexpr int KC   = 64;             // GEMM K-chunk staged per TDM descriptor

__device__ __constant__ int LVL_S[4]     = {100, 50, 25, 13};
__device__ __constant__ int LVL_START[4] = {0, 10000, 12500, 13125};

// ---------------- types ----------------
typedef __attribute__((ext_vector_type(16))) __bf16 v16bf;
typedef __attribute__((ext_vector_type(8)))  __bf16 v8bf;
typedef __attribute__((ext_vector_type(8)))  float  v8f;

__device__ inline unsigned short f2bf(float f) {
  union { float f; unsigned u; } cv; cv.f = f;
  unsigned u = cv.u;
  u += 0x7fffu + ((u >> 16) & 1u);   // round-to-nearest-even
  return (unsigned short)(u >> 16);
}

// ---------------- TDM availability ----------------
#if defined(__has_builtin)
#  if __has_builtin(__builtin_amdgcn_tensor_load_to_lds) && \
      __has_builtin(__builtin_amdgcn_s_wait_tensorcnt)
#    define HAVE_TDM 1
#  endif
#endif
#ifndef HAVE_TDM
#  define HAVE_TDM 0
#endif

#if HAVE_TDM
typedef unsigned int u32x4 __attribute__((ext_vector_type(4)));
typedef int          i32x4 __attribute__((ext_vector_type(4)));
typedef int          i32x8 __attribute__((ext_vector_type(8)));

// 2D tile load: tile = (KC x 64 rows) of 2-byte elements, row stride = strideElems.
// OOB rows (>= tdRows) are zero-filled by the TDM (ISA: reads beyond tensor dim -> 0).
__device__ inline void tdm_load_tile_2d(unsigned ldsOff, const unsigned short* gptr,
                                        unsigned tdCols, unsigned tdRows,
                                        unsigned strideElems)
{
  unsigned long long ga = (unsigned long long)gptr;
  u32x4 g0;
  g0.x = 1u;                                                  // count=1, user mode
  g0.y = ldsOff;                                              // lds_addr
  g0.z = (unsigned)(ga & 0xFFFFFFFFull);                      // global_addr[31:0]
  g0.w = (unsigned)((ga >> 32) & 0x01FFFFFFull) | (2u << 30); // addr[56:32] | type=2
  i32x8 g1;
  g1[0] = (int)(1u << 16);                                    // data_size=2B, mask=0
  g1[1] = (int)((tdCols & 0xFFFFu) << 16);                    // tensor_dim0 lo
  g1[2] = (int)((tdCols >> 16) | ((tdRows & 0xFFFFu) << 16)); // dim0 hi | dim1 lo
  g1[3] = (int)((tdRows >> 16) | ((unsigned)KC << 16));       // dim1 hi | tile_dim0
  g1[4] = 64;                                                 // tile_dim1=64, tile_dim2=0
  g1[5] = (int)strideElems;                                   // tensor_dim0_stride lo
  g1[6] = 0;
  g1[7] = 0;
  i32x4 z4 = {0, 0, 0, 0};
#if __clang_major__ >= 23
  i32x8 z8 = {0, 0, 0, 0, 0, 0, 0, 0};
  __builtin_amdgcn_tensor_load_to_lds(g0, g1, z4, z4, z8, 0);
#else
  __builtin_amdgcn_tensor_load_to_lds(g0, g1, z4, z4, 0);
#endif
}
#endif // HAVE_TDM

// ---------------- fp32 -> bf16 converters ----------------
__global__ __launch_bounds__(256) void cvt_bf16_kernel(
    const float* __restrict__ s, unsigned short* __restrict__ d, long long n) {
  long long i = (long long)blockIdx.x * 256 + threadIdx.x;
  if (i < n) d[i] = f2bf(s[i]);
}

__global__ __launch_bounds__(256) void add_cvt_bf16_kernel(
    const float* __restrict__ a, const float* __restrict__ b,
    unsigned short* __restrict__ d, long long n) {
  long long i = (long long)blockIdx.x * 256 + threadIdx.x;
  if (i < n) d[i] = f2bf(a[i] + b[i]);
}

// ---------------- WMMA GEMM: out[M,N] = A[M,K](bf16) * W[N,K]^T(bf16) + bias -------------
// Block tile 64x64 (8 waves of 16x32). A/W tiles staged in LDS (TDM async, double-buffered),
// fragments fed by ds_load_b128, accumulation via v_wmma_f32_16x16x32_bf16.
__global__ __launch_bounds__(256) void wmma_gemm_bf16(
    const unsigned short* __restrict__ A,   // [M,K] bf16
    const unsigned short* __restrict__ W,   // [N,K] bf16 (row-major weight = B^T)
    const float* __restrict__ bias,         // [N] or nullptr
    float* __restrict__ Cout,               // [M,N] fp32
    int M, int N, int K, int relu)
{
  __shared__ __align__(16) unsigned short Ash[2][64 * KC];
  __shared__ __align__(16) unsigned short Wsh[2][64 * KC];

  const int lane = threadIdx.x & 31;
  const int wid  = threadIdx.x >> 5;
  const int mw   = wid >> 1;               // 0..3  (M sub-tile)
  const int nw   = wid & 1;                // 0..1  (N sub-tile)
  const int nTiles = N >> 6;
  const int row0 = (blockIdx.x / nTiles) << 6;
  const int col0 = (blockIdx.x % nTiles) << 6;
  const int nChunks = K / KC;

#if !HAVE_TDM
  auto stage = [&](int buf, int kk) {
    for (int idx = threadIdx.x; idx < 64 * (KC / 8); idx += 256) {
      int r  = idx >> 3;
      int cc = (idx & 7) * 8;
      int ar = row0 + r; if (ar >= M) ar = M - 1;
      *(uint4*)&Ash[buf][r * KC + cc] = *(const uint4*)&A[(size_t)ar * K + kk + cc];
      *(uint4*)&Wsh[buf][r * KC + cc] = *(const uint4*)&W[(size_t)(col0 + r) * K + kk + cc];
    }
  };
#endif

  // ---- prologue: chunk 0 into buffer 0 ----
#if HAVE_TDM
  if (wid == 0) {
    tdm_load_tile_2d((unsigned)(unsigned long long)(const void*)&Ash[0][0],
                     A + (size_t)row0 * K, (unsigned)K, (unsigned)(M - row0), (unsigned)K);
    tdm_load_tile_2d((unsigned)(unsigned long long)(const void*)&Wsh[0][0],
                     W + (size_t)col0 * K, (unsigned)K, (unsigned)(N - col0), (unsigned)K);
  }
#else
  stage(0, 0);
#endif

  v8f acc0 = {0.f,0.f,0.f,0.f,0.f,0.f,0.f,0.f};
  v8f acc1 = {0.f,0.f,0.f,0.f,0.f,0.f,0.f,0.f};
  const int kh = lane >> 4;

  for (int c = 0; c < nChunks; ++c) {
    const int buf = c & 1;
#if HAVE_TDM
    if (wid == 0) __builtin_amdgcn_s_wait_tensorcnt(0);
    __syncthreads();                        // staged chunk visible to all waves
    if (wid == 0 && c + 1 < nChunks) {      // async-issue next chunk into other buffer
      const int kk = (c + 1) * KC;
      tdm_load_tile_2d((unsigned)(unsigned long long)(const void*)&Ash[buf ^ 1][0],
                       A + (size_t)row0 * K + kk, (unsigned)(K - kk),
                       (unsigned)(M - row0), (unsigned)K);
      tdm_load_tile_2d((unsigned)(unsigned long long)(const void*)&Wsh[buf ^ 1][0],
                       W + (size_t)col0 * K + kk, (unsigned)(K - kk),
                       (unsigned)(N - col0), (unsigned)K);
    }
#else
    __syncthreads();
    if (c + 1 < nChunks) stage(buf ^ 1, (c + 1) * KC);
#endif

    const unsigned short* ar  = &Ash[buf][(mw * 16 + (lane & 15)) * KC];
    const unsigned short* w0p = &Wsh[buf][(nw * 32 +      (lane & 15)) * KC];
    const unsigned short* w1p = &Wsh[buf][(nw * 32 + 16 + (lane & 15)) * KC];
    #pragma unroll
    for (int ks = 0; ks < KC; ks += 32) {
      // A fragment: lane-half selects contiguous K 0..15 / 16..31 (matches B)
      v8bf alo = *(const v8bf*)(ar + ks +     kh * 16);
      v8bf ahi = *(const v8bf*)(ar + ks + 8 + kh * 16);
      v16bf af = __builtin_shufflevector(alo, ahi,
          0,1,2,3,4,5,6,7,8,9,10,11,12,13,14,15);
      // B fragment: lane = column, lane-half selects contiguous K 0..15 / 16..31
      v16bf b0 = *(const v16bf*)(w0p + ks + kh * 16);
      v16bf b1 = *(const v16bf*)(w1p + ks + kh * 16);
      acc0 = __builtin_amdgcn_wmma_f32_16x16x32_bf16(false, af, false, b0,
                                                     (short)0, acc0, false, false);
      acc1 = __builtin_amdgcn_wmma_f32_16x16x32_bf16(false, af, false, b1,
                                                     (short)0, acc1, false, false);
    }
    __syncthreads();                        // all reads of `buf` retired before re-stage
  }

  // C/D layout: VGPR r, lanes<16 -> M=r ; lanes>=16 -> M=8+r ; N = lane&15
  const int n0    = col0 + nw * 32 + (lane & 15);
  const int rbase = row0 + mw * 16 + ((lane >> 4) << 3);
  const float b0v = bias ? bias[n0]      : 0.f;
  const float b1v = bias ? bias[n0 + 16] : 0.f;
  #pragma unroll
  for (int r = 0; r < 8; ++r) {
    int rr = rbase + r;
    if (rr < M) {
      float v0 = acc0[r] + b0v;
      float v1 = acc1[r] + b1v;
      if (relu) { v0 = fmaxf(v0, 0.f); v1 = fmaxf(v1, 0.f); }
      Cout[(size_t)rr * N + n0]      = v0;
      Cout[(size_t)rr * N + n0 + 16] = v1;
    }
  }
}

// ---------------- fused residual-add + LayerNorm (one row of 256 per block) ----------------
__global__ __launch_bounds__(256) void ln_add_kernel(
    const float* resid, const float* delta,
    const float* __restrict__ g, const float* __restrict__ bta, float* out)
{
  __shared__ float s1[256];
  __shared__ float s2[256];
  const int row = blockIdx.x;
  const int c   = threadIdx.x;
  const size_t base = (size_t)row * CDIM;
  float x = resid[base + c] + delta[base + c];
  s1[c] = x; s2[c] = x * x;
  __syncthreads();
  for (int o = 128; o > 0; o >>= 1) {
    if (c < o) { s1[c] += s1[c + o]; s2[c] += s2[c + o]; }
    __syncthreads();
  }
  const float mean = s1[0] * (1.0f / CDIM);
  const float var  = s2[0] * (1.0f / CDIM) - mean * mean;
  const float rs   = rsqrtf(var + 1e-5f);
  out[base + c] = (x - mean) * rs * g[c] + bta[c];
}

// ---------------- encoder reference points ----------------
__global__ __launch_bounds__(256) void enc_ref_kernel(
    const float* __restrict__ vr, float* __restrict__ ref)
{
  int i = blockIdx.x * 256 + threadIdx.x;
  if (i >= ME) return;
  int b = i / LENT, q = i % LENT;
  int lvl = 3;
  if      (q < LVL_START[1]) lvl = 0;
  else if (q < LVL_START[2]) lvl = 1;
  else if (q < LVL_START[3]) lvl = 2;
  int S  = LVL_S[lvl];
  int qq = q - LVL_START[lvl];
  int rr = qq / S, cc = qq % S;
  float bx = (cc + 0.5f) / (vr[(b * 4 + lvl) * 2 + 0] * S);
  float by = (rr + 0.5f) / (vr[(b * 4 + lvl) * 2 + 1] * S);
  #pragma unroll
  for (int t = 0; t < 4; ++t) {
    ref[((size_t)i * 4 + t) * 2 + 0] = bx * vr[(b * 4 + t) * 2 + 0];
    ref[((size_t)i * 4 + t) * 2 + 1] = by * vr[(b * 4 + t) * 2 + 1];
  }
}

// ---------------- decoder reference points: sigmoid(qpos @ ref_w^T + ref_b) * vr ----------------
__global__ void dec_ref_kernel(
    const float* __restrict__ qe, const float* __restrict__ rw,
    const float* __restrict__ rb, const float* __restrict__ vr,
    float* __restrict__ ref)
{
  int i = blockIdx.x * 64 + threadIdx.x;
  if (i >= MD) return;
  int b = i / NQRY, q = i % NQRY;
  float o[2];
  #pragma unroll
  for (int t = 0; t < 2; ++t) {
    float s = rb[t];
    for (int k = 0; k < CDIM; ++k) s += qe[q * 2 * CDIM + k] * rw[t * CDIM + k];
    o[t] = 1.f / (1.f + expf(-s));
  }
  for (int lvl = 0; lvl < 4; ++lvl)
    for (int t = 0; t < 2; ++t)
      ref[((size_t)i * 4 + lvl) * 2 + t] = o[t] * vr[(b * 4 + lvl) * 2 + t];
}

// ---------------- decoder state init / q = out + qpos (bf16) ----------------
__global__ __launch_bounds__(256) void dec_init_kernel(
    const float* __restrict__ qe, float* __restrict__ outS)
{
  int i = blockIdx.x * 256 + threadIdx.x;
  if (i >= MD * CDIM) return;
  int r = i >> 8, c = i & 255, q = r % NQRY;
  outS[i] = qe[q * 2 * CDIM + CDIM + c];
}

__global__ __launch_bounds__(256) void dec_q_cvt_kernel(
    const float* __restrict__ outS, const float* __restrict__ qe,
    unsigned short* __restrict__ d)
{
  int i = blockIdx.x * 256 + threadIdx.x;
  if (i >= MD * CDIM) return;
  int r = i >> 8, c = i & 255, q = r % NQRY;
  d[i] = f2bf(outS[i] + qe[q * 2 * CDIM + c]);
}

// ---------------- MS-deformable attention core: one wave per (b,q,head), lane = channel ----------------
__global__ __launch_bounds__(256) void deform_attn_kernel(
    const float* __restrict__ value,  // [BB, LENT, CDIM]
    const float* __restrict__ off,    // [BB*Q, 256]
    const float* __restrict__ aw,     // [BB*Q, 128]  (logits)
    const float* __restrict__ ref,    // [BB*Q, 4, 2]
    float* __restrict__ out,          // [BB*Q, 256]
    int Q)
{
  const int lane  = threadIdx.x & 31;
  const int wid   = blockIdx.x * 8 + (threadIdx.x >> 5);
  const int total = BB * Q * NHD;
  if (wid >= total) return;
  const int h  = wid % NHD;
  const int bq = wid / NHD;
  const int b  = bq / Q;

  const float* awp = aw + (size_t)bq * (NHD * NLV * NPT) + h * (NLV * NPT);
  float lg[16]; float mx = -1e30f;
  #pragma unroll
  for (int j = 0; j < 16; ++j) { lg[j] = awp[j]; mx = fmaxf(mx, lg[j]); }
  float den = 0.f;
  #pragma unroll
  for (int j = 0; j < 16; ++j) { lg[j] = expf(lg[j] - mx); den += lg[j]; }
  const float inv = 1.f / den;

  const float* offp = off + (size_t)bq * (NHD * NLV * NPT * 2) + h * (NLV * NPT * 2);
  float acc = 0.f;
  #pragma unroll
  for (int lvl = 0; lvl < NLV; ++lvl) {
    const int S = LVL_S[lvl];
    const float* vbase =
        value + ((size_t)b * LENT + LVL_START[lvl]) * CDIM + h * HDIM + lane;
    const float rx = ref[((size_t)bq * 4 + lvl) * 2 + 0];
    const float ry = ref[((size_t)bq * 4 + lvl) * 2 + 1];
    #pragma unroll
    for (int p = 0; p < NPT; ++p) {
      const float ox = offp[lvl * 8 + p * 2 + 0];
      const float oy = offp[lvl * 8 + p * 2 + 1];
      const float x = (rx + ox / (float)S) * S - 0.5f;
      const float y = (ry + oy / (float)S) * S - 0.5f;
      const float x0f = floorf(x), y0f = floorf(y);
      const int x0 = (int)x0f, y0 = (int)y0f;
      const float wx = x - x0f, wy = y - y0f;
      const float w  = lg[lvl * NPT + p] * inv;
      float sv[4];
      const int xs[2] = {x0, x0 + 1};
      const int ys[2] = {y0, y0 + 1};
      #pragma unroll
      for (int cy = 0; cy < 2; ++cy)
        #pragma unroll
        for (int cx = 0; cx < 2; ++cx) {
          int xi = xs[cx], yi = ys[cy];
          bool valid = (xi >= 0) && (xi < S) && (yi >= 0) && (yi < S);
          int xc = xi < 0 ? 0 : (xi >= S ? S - 1 : xi);
          int yc = yi < 0 ? 0 : (yi >= S ? S - 1 : yi);
          float v = vbase[(size_t)(yc * S + xc) * CDIM];
          sv[cy * 2 + cx] = valid ? v : 0.f;
        }
      acc += w * ((1.f - wy) * ((1.f - wx) * sv[0] + wx * sv[1]) +
                  wy        * ((1.f - wx) * sv[2] + wx * sv[3]));
    }
  }
  out[(size_t)bq * CDIM + h * HDIM + lane] = acc;
}

// ---------------- decoder self-attention: one block per (b,head), K staged in LDS ----------------
__global__ __launch_bounds__(256) void mha_kernel(
    const float* __restrict__ qk,   // [MD, 512]: cols 0..255 = Q proj, 256..511 = K proj
    const float* __restrict__ v,    // [MD, 256]
    float* __restrict__ o)          // [MD, 256]
{
  __shared__ float Ksh[NQRY * HDIM];       // 38400 B
  const int b = blockIdx.x / NHD;
  const int h = blockIdx.x % NHD;

  for (int i = threadIdx.x; i < NQRY * HDIM; i += 256) {
    int r = i >> 5, c = i & 31;
    Ksh[i] = qk[((size_t)(b * NQRY) + r) * 512 + 256 + h * HDIM + c];
  }
  __syncthreads();

  const float scale = 0.1767766952966369f;   // 1/sqrt(32)
  for (int q = threadIdx.x; q < NQRY; q += 256) {
    float qv[32];
    const float* qp = qk + ((size_t)(b * NQRY) + q) * 512 + h * HDIM;
    #pragma unroll
    for (int c = 0; c < 32; ++c) qv[c] = qp[c];

    float mx = -1e30f;
    for (int k = 0; k < NQRY; ++k) {
      const float* kp = &Ksh[k * 32];
      float s = 0.f;
      #pragma unroll
      for (int c = 0; c < 32; ++c) s += qv[c] * kp[c];
      s *= scale;
      if (s > mx) mx = s;
    }
    float acc[32];
    #pragma unroll
    for (int c = 0; c < 32; ++c) acc[c] = 0.f;
    float den = 0.f;
    for (int k = 0; k < NQRY; ++k) {
      const float* kp = &Ksh[k * 32];
      float s = 0.f;
      #pragma unroll
      for (int c = 0; c < 32; ++c) s += qv[c] * kp[c];
      float e = expf(s * scale - mx);
      den += e;
      const float* vp = v + ((size_t)(b * NQRY) + k) * 256 + h * HDIM;
      #pragma unroll
      for (int c = 0; c < 32; ++c) acc[c] += e * vp[c];
    }
    const float invd = 1.f / den;
    float* op = o + ((size_t)(b * NQRY) + q) * 256 + h * HDIM;
    #pragma unroll
    for (int c = 0; c < 32; ++c) op[c] = acc[c] * invd;
  }
}

// ---------------- host-side helpers ----------------
static void gemm_launch(const unsigned short* A, const unsigned short* W,
                        const float* bias, float* out, int M, int N, int K,
                        int relu, hipStream_t s) {
  int blocks = ((M + 63) / 64) * (N / 64);
  wmma_gemm_bf16<<<blocks, 256, 0, s>>>(A, W, bias, out, M, N, K, relu);
}
static void cvt(const float* s, unsigned short* d, long long n, hipStream_t st) {
  cvt_bf16_kernel<<<(unsigned)((n + 255) / 256), 256, 0, st>>>(s, d, n);
}
static void add_cvt(const float* a, const float* b, unsigned short* d,
                    long long n, hipStream_t st) {
  add_cvt_bf16_kernel<<<(unsigned)((n + 255) / 256), 256, 0, st>>>(a, b, d, n);
}

extern "C" void kernel_launch(void* const* d_in, const int* in_sizes, int n_in,
                              void* d_out, int out_size, void* d_ws, size_t ws_size,
                              hipStream_t stream)
{
  (void)in_sizes; (void)n_in; (void)out_size; (void)ws_size;
  const float* src  = (const float*)d_in[0];
  const float* pos  = (const float*)d_in[1];
  const float* qe   = (const float*)d_in[2];
  const float* vr   = (const float*)d_in[3];
  const float* e_off_w = (const float*)d_in[4];   const float* e_off_b = (const float*)d_in[5];
  const float* e_aw_w  = (const float*)d_in[6];   const float* e_aw_b  = (const float*)d_in[7];
  const float* e_val_w = (const float*)d_in[8];   const float* e_val_b = (const float*)d_in[9];
  const float* e_out_w = (const float*)d_in[10];  const float* e_out_b = (const float*)d_in[11];
  const float* e_f1_w  = (const float*)d_in[12];  const float* e_f1_b  = (const float*)d_in[13];
  const float* e_f2_w  = (const float*)d_in[14];  const float* e_f2_b  = (const float*)d_in[15];
  const float* d_sa_in_w  = (const float*)d_in[16]; const float* d_sa_in_b  = (const float*)d_in[17];
  const float* d_sa_out_w = (const float*)d_in[18]; const float* d_sa_out_b = (const float*)d_in[19];
  const float* d_off_w = (const float*)d_in[20];  const float* d_off_b = (const float*)d_in[21];
  const float* d_aw_w  = (const float*)d_in[22];  const float* d_aw_b  = (const float*)d_in[23];
  const float* d_val_w = (const float*)d_in[24];  const float* d_val_b = (const float*)d_in[25];
  const float* d_op_w  = (const float*)d_in[26];  const float* d_op_b  = (const float*)d_in[27];
  const float* d_f1_w  = (const float*)d_in[28];  const float* d_f1_b  = (const float*)d_in[29];
  const float* d_f2_w  = (const float*)d_in[30];  const float* d_f2_b  = (const float*)d_in[31];
  const float* ref_w   = (const float*)d_in[32];  const float* ref_b   = (const float*)d_in[33];
  const float* e_ln1_g = (const float*)d_in[34];  const float* e_ln1_b = (const float*)d_in[35];
  const float* e_ln2_g = (const float*)d_in[36];  const float* e_ln2_b = (const float*)d_in[37];
  const float* d_ln1_g = (const float*)d_in[38];  const float* d_ln1_b = (const float*)d_in[39];
  const float* d_ln2_g = (const float*)d_in[40];  const float* d_ln2_b = (const float*)d_in[41];
  const float* d_ln3_g = (const float*)d_in[42];  const float* d_ln3_b = (const float*)d_in[43];

  size_t woff = 0;
  auto alloc  = [&](size_t bytes) -> void* {
    void* r = (char*)d_ws + woff;
    woff += (bytes + 255) & ~(size_t)255;
    return r;
  };
  auto allocF = [&](size_t n) { return (float*)alloc(n * 4); };
  auto allocH = [&](size_t n) { return (unsigned short*)alloc(n * 2); };

  // ---- bf16 weight copies (re-done every call; deterministic) ----
  unsigned short* w_eoff  = allocH((size_t)NLAY * 256 * CDIM);
  unsigned short* w_eaw   = allocH((size_t)NLAY * 128 * CDIM);
  unsigned short* w_eval  = allocH((size_t)NLAY * 256 * CDIM);
  unsigned short* w_eout  = allocH((size_t)NLAY * 256 * CDIM);
  unsigned short* w_ef1   = allocH((size_t)NLAY * FFD * CDIM);
  unsigned short* w_ef2   = allocH((size_t)NLAY * CDIM * FFD);
  unsigned short* w_dsain = allocH((size_t)NLAY * 768 * CDIM);
  unsigned short* w_dsaout= allocH((size_t)NLAY * 256 * CDIM);
  unsigned short* w_doff  = allocH((size_t)NLAY * 256 * CDIM);
  unsigned short* w_daw   = allocH((size_t)NLAY * 128 * CDIM);
  unsigned short* w_dval  = allocH((size_t)NLAY * 256 * CDIM);
  unsigned short* w_dout  = allocH((size_t)NLAY * 256 * CDIM);
  unsigned short* w_df1   = allocH((size_t)NLAY * FFD * CDIM);
  unsigned short* w_df2   = allocH((size_t)NLAY * CDIM * FFD);

  cvt(e_off_w,  w_eoff,  (long long)NLAY * 256 * CDIM, stream);
  cvt(e_aw_w,   w_eaw,   (long long)NLAY * 128 * CDIM, stream);
  cvt(e_val_w,  w_eval,  (long long)NLAY * 256 * CDIM, stream);
  cvt(e_out_w,  w_eout,  (long long)NLAY * 256 * CDIM, stream);
  cvt(e_f1_w,   w_ef1,   (long long)NLAY * FFD * CDIM, stream);
  cvt(e_f2_w,   w_ef2,   (long long)NLAY * CDIM * FFD, stream);
  cvt(d_sa_in_w, w_dsain, (long long)NLAY * 768 * CDIM, stream);
  cvt(d_sa_out_w,w_dsaout,(long long)NLAY * 256 * CDIM, stream);
  cvt(d_off_w,  w_doff,  (long long)NLAY * 256 * CDIM, stream);
  cvt(d_aw_w,   w_daw,   (long long)NLAY * 128 * CDIM, stream);
  cvt(d_val_w,  w_dval,  (long long)NLAY * 256 * CDIM, stream);
  cvt(d_op_w,   w_dout,  (long long)NLAY * 256 * CDIM, stream);
  cvt(d_f1_w,   w_df1,   (long long)NLAY * FFD * CDIM, stream);
  cvt(d_f2_w,   w_df2,   (long long)NLAY * CDIM * FFD, stream);

  // ---- activations ----
  float* mem   = allocF((size_t)ME * CDIM);
  float* value = allocF((size_t)ME * CDIM);
  float* offb  = allocF((size_t)ME * CDIM);
  float* awb   = allocF((size_t)ME * 128);
  float* samp  = allocF((size_t)ME * CDIM);
  float* abuf  = allocF((size_t)ME * CDIM);
  float* hbuf  = allocF((size_t)ME * FFD);
  float* refE  = allocF((size_t)ME * 8);
  unsigned short* qbf   = allocH((size_t)ME * CDIM);   // also reused for samp-bf16
  unsigned short* membf = allocH((size_t)ME * CDIM);
  unsigned short* hbf   = allocH((size_t)ME * FFD);

  float* doutS = allocF((size_t)MD * CDIM);
  float* dqk   = allocF((size_t)MD * 512);
  float* dv    = allocF((size_t)MD * CDIM);
  float* d_o   = allocF((size_t)MD * CDIM);
  float* dsa   = allocF((size_t)MD * CDIM);
  float* doff  = allocF((size_t)MD * CDIM);
  float* daw   = allocF((size_t)MD * 128);
  float* dsamp = allocF((size_t)MD * CDIM);
  float* dca   = allocF((size_t)MD * CDIM);
  float* dh    = allocF((size_t)MD * FFD);
  float* dff   = allocF((size_t)MD * CDIM);
  float* refD  = allocF((size_t)MD * 8);
  unsigned short* dqbf = allocH((size_t)MD * CDIM);
  unsigned short* dobf = allocH((size_t)MD * CDIM);
  unsigned short* dhbf = allocH((size_t)MD * FFD);

  // ================= encoder =================
  hipMemcpyAsync(mem, src, sizeof(float) * (size_t)ME * CDIM,
                 hipMemcpyDeviceToDevice, stream);
  enc_ref_kernel<<<(ME + 255) / 256, 256, 0, stream>>>(vr, refE);

  for (int i = 0; i < NLAY; ++i) {
    add_cvt(mem, pos, qbf, (long long)ME * CDIM, stream);        // q = mem + pos
    cvt(mem, membf, (long long)ME * CDIM, stream);
    gemm_launch(membf, w_eval + (size_t)i * 256 * CDIM, e_val_b + i * 256, value, ME, 256, CDIM, 0, stream);
    gemm_launch(qbf,   w_eoff + (size_t)i * 256 * CDIM, e_off_b + i * 256, offb,  ME, 256, CDIM, 0, stream);
    gemm_launch(qbf,   w_eaw  + (size_t)i * 128 * CDIM, e_aw_b  + i * 128, awb,   ME, 128, CDIM, 0, stream);
    deform_attn_kernel<<<(BB * LENT * NHD + 7) / 8, 256, 0, stream>>>(value, offb, awb, refE, samp, LENT);
    cvt(samp, qbf, (long long)ME * CDIM, stream);
    gemm_launch(qbf, w_eout + (size_t)i * 256 * CDIM, e_out_b + i * 256, abuf, ME, 256, CDIM, 0, stream);
    ln_add_kernel<<<ME, 256, 0, stream>>>(mem, abuf, e_ln1_g + i * CDIM, e_ln1_b + i * CDIM, mem);
    cvt(mem, membf, (long long)ME * CDIM, stream);
    gemm_launch(membf, w_ef1 + (size_t)i * FFD * CDIM, e_f1_b + i * FFD, hbuf, ME, FFD, CDIM, 1, stream);
    cvt(hbuf, hbf, (long long)ME * FFD, stream);
    gemm_launch(hbf, w_ef2 + (size_t)i * CDIM * FFD, e_f2_b + i * CDIM, abuf, ME, CDIM, FFD, 0, stream);
    ln_add_kernel<<<ME, 256, 0, stream>>>(mem, abuf, e_ln2_g + i * CDIM, e_ln2_b + i * CDIM, mem);
  }

  // ================= decoder =================
  dec_init_kernel<<<(MD * CDIM + 255) / 256, 256, 0, stream>>>(qe, doutS);
  dec_ref_kernel<<<(MD + 63) / 64, 64, 0, stream>>>(qe, ref_w, ref_b, vr, refD);
  cvt(mem, membf, (long long)ME * CDIM, stream);   // final memory, reused by all layers

  for (int i = 0; i < NLAY; ++i) {
    // --- self attention: q/k from out+qpos, v from out ---
    dec_q_cvt_kernel<<<(MD * CDIM + 255) / 256, 256, 0, stream>>>(doutS, qe, dqbf);
    cvt(doutS, dobf, (long long)MD * CDIM, stream);
    gemm_launch(dqbf, w_dsain + (size_t)i * 768 * CDIM,                     d_sa_in_b + i * 768,       dqk, MD, 512, CDIM, 0, stream);
    gemm_launch(dobf, w_dsain + (size_t)i * 768 * CDIM + (size_t)512*CDIM, d_sa_in_b + i * 768 + 512, dv,  MD, 256, CDIM, 0, stream);
    mha_kernel<<<BB * NHD, 256, 0, stream>>>(dqk, dv, d_o);
    cvt(d_o, dobf, (long long)MD * CDIM, stream);
    gemm_launch(dobf, w_dsaout + (size_t)i * 256 * CDIM, d_sa_out_b + i * 256, dsa, MD, 256, CDIM, 0, stream);
    ln_add_kernel<<<MD, 256, 0, stream>>>(doutS, dsa, d_ln2_g + i * CDIM, d_ln2_b + i * CDIM, doutS);

    // --- cross deformable attention over encoder memory ---
    dec_q_cvt_kernel<<<(MD * CDIM + 255) / 256, 256, 0, stream>>>(doutS, qe, dqbf);
    gemm_launch(membf, w_dval + (size_t)i * 256 * CDIM, d_val_b + i * 256, value, ME, 256, CDIM, 0, stream);
    gemm_launch(dqbf,  w_doff + (size_t)i * 256 * CDIM, d_off_b + i * 256, doff,  MD, 256, CDIM, 0, stream);
    gemm_launch(dqbf,  w_daw  + (size_t)i * 128 * CDIM, d_aw_b  + i * 128, daw,   MD, 128, CDIM, 0, stream);
    deform_attn_kernel<<<(BB * NQRY * NHD + 7) / 8, 256, 0, stream>>>(value, doff, daw, refD, dsamp, NQRY);
    cvt(dsamp, dobf, (long long)MD * CDIM, stream);
    gemm_launch(dobf, w_dout + (size_t)i * 256 * CDIM, d_op_b + i * 256, dca, MD, 256, CDIM, 0, stream);
    ln_add_kernel<<<MD, 256, 0, stream>>>(doutS, dca, d_ln1_g + i * CDIM, d_ln1_b + i * CDIM, doutS);

    // --- FFN ---
    cvt(doutS, dobf, (long long)MD * CDIM, stream);
    gemm_launch(dobf, w_df1 + (size_t)i * FFD * CDIM, d_f1_b + i * FFD, dh, MD, FFD, CDIM, 1, stream);
    cvt(dh, dhbf, (long long)MD * FFD, stream);
    gemm_launch(dhbf, w_df2 + (size_t)i * CDIM * FFD, d_f2_b + i * CDIM, dff, MD, CDIM, FFD, 0, stream);
    float* tgt = (i == NLAY - 1) ? (float*)d_out : doutS;
    ln_add_kernel<<<MD, 256, 0, stream>>>(doutS, dff, d_ln3_g + i * CDIM, d_ln3_b + i * CDIM, tgt);
  }
}
